// CRF_27530740367390
// MI455X (gfx1250) — compile-verified, matrix operand order checked by reading, not run on hardware
//
#include <hip/hip_runtime.h>
#include <math.h>

#define NUM_TAGS 54
#define START_TAG 52
#define STOP_TAG 53
#define SEQ_LEN 512
#define BATCH 1024
#define NEG_INF_V (-10000.0f)
#define ALPHA_STRIDE 68            // padded row stride (floats): bank-conflict-free, 16B aligned
#define DUMMY_BASE (16 * ALPHA_STRIDE)   // sink row for masked / out-of-range stores
#define TILE_ELEMS (16 * NUM_TAGS)       // 864 floats: one step's features for 16 batches

typedef __attribute__((ext_vector_type(16))) _Float16 v16h;
typedef __attribute__((ext_vector_type(8)))  float    v8f;
typedef __attribute__((ext_vector_type(4)))  unsigned int u32x4;
typedef __attribute__((ext_vector_type(8)))  int       i32x8;
typedef __attribute__((ext_vector_type(4)))  int       i32x4;

// raw v_log_f32 (log2) * ln2 — avoids ocml denormal-fixup sequence
static __device__ __forceinline__ float fast_log(float x) {
#if __has_builtin(__builtin_amdgcn_logf)
    return __builtin_amdgcn_logf(x) * 0.6931471805599453f;
#else
    float r;
    asm volatile("v_log_f32_e32 %0, %1" : "=v"(r) : "v"(x));
    return r * 0.6931471805599453f;
#endif
}

// ---------------------------------------------------------------------------
// TDM: DMA one contiguous 864-float (3456 B) feature tile from global -> LDS.
// D# per CDNA5 ISA ch.8: group0 = {flags, lds_addr, global_addr, type=2},
// group1 = {data_size=4B, tensor_dim0=tile_dim0=864, dim1=1, stride0=864},
// groups 2/3 zeroed (<=2D tensor). Tracked with TENSORcnt.
// 6-arg builtin form (clang-23 / therock headers).
// ---------------------------------------------------------------------------
static __device__ __forceinline__ void tdm_load_tile(const float* src, unsigned lds_off) {
    unsigned long long ga = (unsigned long long)(uintptr_t)src;
    u32x4 g0;
    g0[0] = 1u;                                   // count=1, user descriptor
    g0[1] = lds_off;                              // LDS byte address of dest tile
    g0[2] = (unsigned)(ga & 0xFFFFFFFFu);         // global_addr[31:0]
    g0[3] = (unsigned)((ga >> 32) & 0x01FFFFFFu)  // global_addr[56:32]
          | (2u << 30);                           // type = 2 ("image")
    i32x8 g1;
    g1[0] = (int)(2u << 16);                      // data_size = 2 -> 4 bytes
    g1[1] = (int)(((unsigned)TILE_ELEMS & 0xFFFFu) << 16);  // tensor_dim0[15:0]
    g1[2] = (int)(((unsigned)TILE_ELEMS >> 16) | (1u << 16)); // dim0[31:16], tensor_dim1=1
    g1[3] = (int)((unsigned)TILE_ELEMS << 16);    // tile_dim0 = 864
    g1[4] = 0;                                    // tile_dim1 = tile_dim2 = 0 (1D)
    g1[5] = (int)TILE_ELEMS;                      // tensor_dim0_stride[31:0]
    g1[6] = 0;
    g1[7] = 0;
    i32x4 z4 = {0, 0, 0, 0};
    i32x8 z8 = {0, 0, 0, 0, 0, 0, 0, 0};
    __builtin_amdgcn_tensor_load_to_lds(g0, g1, z4, z4, z8, 0);
}

// ---------------------------------------------------------------------------
// Kernel 1: persistent CRF forward recursion.
// Grid: 64 blocks x 128 threads (4 waves). Block owns 16 batches.
// Wave w owns output-tag tile n in [16w, 16w+16). K (old tag) padded 54->64.
// alpha state lives in LDS; W = exp(trans) lives in VGPRs as WMMA B operands.
// Emission tiles are double-buffered in LDS via the Tensor Data Mover.
// ---------------------------------------------------------------------------
__launch_bounds__(128)
__global__ void crf_forward_kernel(const float* __restrict__ em,
                                   const float* __restrict__ trans,
                                   const int*   __restrict__ mask,
                                   float*       __restrict__ fwd_out)
{
    __shared__ __align__(16) float alpha[16 * ALPHA_STRIDE + 32]; // +32 = dummy sink
    __shared__ __align__(16) float feat_lds[2][TILE_ELEMS];      // TDM double buffer
    __shared__ __align__(16) float mxbuf[16];
    __shared__ __align__(16) int   mkbuf[16];

    const int tid  = threadIdx.x;
    const int w    = tid >> 5;        // wave id == N tile id
    const int lane = tid & 31;
    const int g    = lane >> 4;       // lane half-group
    const int nl   = lane & 15;       // row index within tile (M for A, N for B/D)
    const int batch_base = blockIdx.x * 16;

    const unsigned feat_off[2] = {
        (unsigned)(uintptr_t)&feat_lds[0][0],
        (unsigned)(uintptr_t)&feat_lds[1][0]
    };
    const float* em_blk = em + (size_t)batch_base * NUM_TAGS;   // step stride = B*54

    // ---- init alpha: NEG_INF everywhere, 0 at START; padding cols stay NEG_INF
    for (int i = tid; i < 16 * ALPHA_STRIDE + 32; i += 128) {
        int k = i % ALPHA_STRIDE;
        alpha[i] = (k == START_TAG && i < 16 * ALPHA_STRIDE) ? 0.0f : NEG_INF_V;
    }

    // ---- kick off DMA of step-0 feature tile
    if (w == 0) tdm_load_tile(em_blk, feat_off[0]);

    // ---- build B operands: W[k][n] = exp(trans[n][k]), two K tiles (k<32, k>=32)
    // B^T stored in A-layout: lane row n = 16w+nl, half e -> k per ISA table.
    v16h Wk0, Wk1;
    {
        const int n = 16 * w + nl;
        #pragma unroll
        for (int e = 0; e < 16; ++e) {
            int kk = (e < 8) ? (g * 8 + e) : (16 + g * 8 + (e - 8));
            float v0 = 0.0f, v1 = 0.0f;
            if (n < NUM_TAGS) {
                if (kk      < NUM_TAGS) v0 = __expf(trans[n * NUM_TAGS + kk]);
                if (kk + 32 < NUM_TAGS) v1 = __expf(trans[n * NUM_TAGS + kk + 32]);
            }
            Wk0[e] = (_Float16)v0;
            Wk1[e] = (_Float16)v1;
        }
    }
    __syncthreads();

    const int n_out = 16 * w + nl;                            // output tag column
    const int nc    = (n_out < NUM_TAGS) ? n_out : (NUM_TAGS - 1); // clamped for reads

    for (int s = 0; s < SEQ_LEN; ++s) {
        // ---- fused phase 2+3: lane reads only its 32 k-values (8 float4),
        //      partial max + cross-half-group shuffle gives full-row max,
        //      then exp() on the already-loaded values builds the A operands.
        const float4* arow4 = (const float4*)(&alpha[nl * ALPHA_STRIDE]);
        float4 c0 = arow4[2 * g + 0],  c1 = arow4[2 * g + 1];   // k = 8g .. 8g+7
        float4 c2 = arow4[4 + 2 * g],  c3 = arow4[5 + 2 * g];   // k = 16+8g ..
        float4 c4 = arow4[8 + 2 * g],  c5 = arow4[9 + 2 * g];   // k = 32+8g ..
        float4 c6 = arow4[12 + 2 * g], c7 = arow4[13 + 2 * g];  // k = 48+8g ..

        float lm = fmaxf(fmaxf(fmaxf(c0.x, c0.y), fmaxf(c0.z, c0.w)),
                         fmaxf(fmaxf(c1.x, c1.y), fmaxf(c1.z, c1.w)));
        lm = fmaxf(lm, fmaxf(fmaxf(fmaxf(c2.x, c2.y), fmaxf(c2.z, c2.w)),
                             fmaxf(fmaxf(c3.x, c3.y), fmaxf(c3.z, c3.w))));
        lm = fmaxf(lm, fmaxf(fmaxf(fmaxf(c4.x, c4.y), fmaxf(c4.z, c4.w)),
                             fmaxf(fmaxf(c5.x, c5.y), fmaxf(c5.z, c5.w))));
        lm = fmaxf(lm, fmaxf(fmaxf(fmaxf(c6.x, c6.y), fmaxf(c6.z, c6.w)),
                             fmaxf(fmaxf(c7.x, c7.y), fmaxf(c7.z, c7.w))));
        const float mx = fmaxf(lm, __shfl_xor(lm, 16, 32));    // full row max

        if (tid < 16) {                       // one writer per m; g==0 lanes
            mxbuf[tid] = mx;
            mkbuf[tid] = mask[s * BATCH + batch_base + tid];
        }

        #define EH(x) ((_Float16)__expf((x) - mx))
        v16h A0 = { EH(c0.x), EH(c0.y), EH(c0.z), EH(c0.w),
                    EH(c1.x), EH(c1.y), EH(c1.z), EH(c1.w),
                    EH(c2.x), EH(c2.y), EH(c2.z), EH(c2.w),
                    EH(c3.x), EH(c3.y), EH(c3.z), EH(c3.w) };
        v16h A1 = { EH(c4.x), EH(c4.y), EH(c4.z), EH(c4.w),
                    EH(c5.x), EH(c5.y), EH(c5.z), EH(c5.w),
                    EH(c6.x), EH(c6.y), EH(c6.z), EH(c6.w),
                    EH(c7.x), EH(c7.y), EH(c7.z), EH(c7.w) };
        #undef EH

        // ---- TDM pipeline (wave 0): launch DMA for step s+1, then ensure the
        //      step-s tile is resident (TDM completes in-order -> cnt<=1 is enough).
        if (w == 0) {
            if (s + 1 < SEQ_LEN) {
                tdm_load_tile(em_blk + (size_t)(s + 1) * BATCH * NUM_TAGS,
                              feat_off[(s + 1) & 1]);
                __builtin_amdgcn_s_wait_tensorcnt(1);
            } else {
                __builtin_amdgcn_s_wait_tensorcnt(0);
            }
        }
        __syncthreads();   // barrier #1: alpha reads done, mx/mask + feat tile visible

        // ---- phase 4: WMMA, uniform control flow (EXEC all ones)
        v8f acc = {};
        acc = __builtin_amdgcn_wmma_f32_16x16x32_f16(false, A0, false, Wk0,
                                                     (short)0, acc, false, false);
        acc = __builtin_amdgcn_wmma_f32_16x16x32_f16(false, A1, false, Wk1,
                                                     (short)0, acc, false, false);

        // ---- phase 5 (branchless): feature reads from the DMA-staged LDS tile,
        //      masked/out-of-range lanes redirect their store into the dummy sink.
        const float* fb = feat_lds[s & 1];
        float feat[8];
        #pragma unroll
        for (int r = 0; r < 8; ++r)
            feat[r] = fb[(r + 8 * g) * NUM_TAGS + nc];

        float4 mxa = ((const float4*)mxbuf)[2 * g];
        float4 mxb = ((const float4*)mxbuf)[2 * g + 1];
        int4   mka = ((const int4*)mkbuf)[2 * g];
        int4   mkb = ((const int4*)mkbuf)[2 * g + 1];
        float mxv[8] = { mxa.x, mxa.y, mxa.z, mxa.w, mxb.x, mxb.y, mxb.z, mxb.w };
        int   mkv[8] = { mka.x, mka.y, mka.z, mka.w, mkb.x, mkb.y, mkb.z, mkb.w };

        #pragma unroll
        for (int r = 0; r < 8; ++r) {
            int m  = r + 8 * g;                  // D layout: VGPR r holds M=r / r+8
            float na = fast_log(acc[r]) + mxv[r] + feat[r];
            bool valid = (n_out < NUM_TAGS) && (mkv[r] > 0);
            int addr = valid ? (m * ALPHA_STRIDE + n_out) : (DUMMY_BASE + lane);
            alpha[addr] = na;                    // unconditional store, no divergence
        }
        __syncthreads();   // barrier #2: writes visible before next-step reads
    }

    // ---- final forward score: logsumexp(alpha + trans[STOP, :]) per batch row
    if (tid < 16) {
        const float* ar = &alpha[tid * ALPHA_STRIDE];
        float vmax = -3.0e38f;
        for (int nn = 0; nn < NUM_TAGS; ++nn) {
            float v = ar[nn] + trans[STOP_TAG * NUM_TAGS + nn];
            vmax = fmaxf(vmax, v);
        }
        float ssum = 0.0f;
        for (int nn = 0; nn < NUM_TAGS; ++nn) {
            float v = ar[nn] + trans[STOP_TAG * NUM_TAGS + nn];
            ssum += __expf(v - vmax);
        }
        fwd_out[batch_base + tid] = vmax + fast_log(ssum);
    }
}

// ---------------------------------------------------------------------------
// Kernel 2: gold-path score per batch + per-batch loss and mask length.
// ---------------------------------------------------------------------------
__global__ void crf_gold_kernel(const float* __restrict__ em,
                                const float* __restrict__ trans,
                                const int*   __restrict__ tags,
                                const int*   __restrict__ mask,
                                const float* __restrict__ fwd,
                                float*       __restrict__ loss,
                                float*       __restrict__ lenf)
{
    int b = blockIdx.x * blockDim.x + threadIdx.x;
    if (b >= BATCH) return;

    int prev = tags[b];
    float gold = trans[prev * NUM_TAGS + START_TAG] + em[(size_t)b * NUM_TAGS + prev];
    int len = mask[b];

    for (int s = 1; s < SEQ_LEN; ++s) {
        int tg = tags[s * BATCH + b];
        int mv = mask[s * BATCH + b];
        float mvf = (float)mv;
        gold += mvf * (trans[tg * NUM_TAGS + prev] +
                       em[((size_t)s * BATCH + b) * NUM_TAGS + tg]);
        len += mv;
        prev = tg;
    }
    int last = tags[(size_t)(len - 1) * BATCH + b];
    gold += trans[STOP_TAG * NUM_TAGS + last];

    loss[b] = fwd[b] - gold;
    lenf[b] = (float)len;
}

// ---------------------------------------------------------------------------
// Kernel 3: deterministic fixed-order reduction -> scalar loss.
// ---------------------------------------------------------------------------
__global__ void crf_reduce_kernel(const float* __restrict__ loss,
                                  const float* __restrict__ lenf,
                                  float*       __restrict__ out)
{
    __shared__ float sl[256];
    __shared__ float sm[256];
    int t = threadIdx.x;
    float a = 0.0f, c = 0.0f;
    for (int i = t; i < BATCH; i += 256) { a += loss[i]; c += lenf[i]; }
    sl[t] = a; sm[t] = c;
    __syncthreads();
    for (int off = 128; off > 0; off >>= 1) {
        if (t < off) { sl[t] += sl[t + off]; sm[t] += sm[t + off]; }
        __syncthreads();
    }
    if (t == 0) out[0] = sl[0] / sm[0];
}

extern "C" void kernel_launch(void* const* d_in, const int* in_sizes, int n_in,
                              void* d_out, int out_size, void* d_ws, size_t ws_size,
                              hipStream_t stream) {
    const float* em    = (const float*)d_in[0];  // (512,1024,54) f32
    const float* trans = (const float*)d_in[1];  // (54,54) f32
    const int*   tags  = (const int*)d_in[2];    // (512,1024) int
    const int*   mask  = (const int*)d_in[3];    // (512,1024) int
    float* out = (float*)d_out;

    float* ws   = (float*)d_ws;
    float* fwd  = ws;           // [1024]
    float* loss = ws + 1024;    // [1024]
    float* lenf = ws + 2048;    // [1024]

    crf_forward_kernel<<<BATCH / 16, 128, 0, stream>>>(em, trans, mask, fwd);
    crf_gold_kernel<<<BATCH / 256, 256, 0, stream>>>(em, trans, tags, mask, fwd, loss, lenf);
    crf_reduce_kernel<<<1, 256, 0, stream>>>(loss, lenf, out);
}
